// ScannedRNN_80882824118579
// MI455X (gfx1250) — matrix operand order repaired and checked by reading, not compiled, not run on hardware
//
#include <hip/hip_runtime.h>
#include <hip/hip_bf16.h>

// LSTM scan: T=512, B=256, D=128, H=256.
// 16 independent workgroups (one per 16-row batch tile), each runs the whole
// 512-step scan. BF16 WMMA 16x16x32, fp32 accumulation. x tile and h tile are
// staged in LDS as bf16; weights pre-transposed to bf16 in workspace so each
// lane's B-fragment is two 16B loads from one row. obs/ys use non-temporal path so
// weights stay resident in L2.

typedef __attribute__((ext_vector_type(16))) __bf16 bf16x16;
typedef __attribute__((ext_vector_type(8)))  float  f32x8;
typedef __attribute__((ext_vector_type(4)))  float  f32x4v;
typedef __attribute__((ext_vector_type(4)))  __bf16 bf16x4v;

#define T_DIM 512
#define B_DIM 256
#define D_DIM 128
#define H_DIM 256
#define FH_DIM 1024          // 4*H
#define BT 16                // batch tile rows per workgroup
#define HPITCH 264           // H_DIM + 8 bf16 pad (16B-aligned rows, bank-spread)
#define XPITCH 136           // D_DIM + 8 bf16 pad

static __device__ __forceinline__ __bf16 f2bf(float f) {
    return (__bf16)f;        // hardware v_cvt (RNE) on gfx1250
}

static __device__ __forceinline__ float sigmoidf_(float x) {
    return 1.0f / (1.0f + __expf(-x));
}
static __device__ __forceinline__ float tanhf_(float x) {
    float e = __expf(-2.0f * x);
    return (1.0f - e) / (1.0f + e);
}

union AFrag { bf16x16 v; int4 q[2]; };

// One-time weight conversion: Wx (D x 4H) -> WxT bf16 (4H x D),
//                             Wh (H x 4H) -> WhT bf16 (4H x H).
__global__ void cvt_weights(const float* __restrict__ Wx, const float* __restrict__ Wh,
                            __bf16* __restrict__ WxT, __bf16* __restrict__ WhT) {
    int idx = blockIdx.x * blockDim.x + threadIdx.x;
    const int NX = FH_DIM * D_DIM;
    if (idx < NX) {
        int n = idx / D_DIM, k = idx - n * D_DIM;
        WxT[idx] = f2bf(Wx[k * FH_DIM + n]);
    } else {
        int j = idx - NX;
        if (j < FH_DIM * H_DIM) {
            int n = j / H_DIM, k = j - n * H_DIM;
            WhT[j] = f2bf(Wh[k * FH_DIM + n]);
        }
    }
}

__global__ __launch_bounds__(512, 1)
void lstm_scan(const float* __restrict__ obs, const int* __restrict__ reset,
               const float* __restrict__ c0, const float* __restrict__ h0,
               const float* __restrict__ bias,
               const __bf16* __restrict__ WxT, const __bf16* __restrict__ WhT,
               float* __restrict__ out) {
    __shared__ __align__(16) __bf16 h_lds[BT * HPITCH];
    __shared__ __align__(16) __bf16 x_lds[BT * XPITCH];
    __shared__ __align__(16) float  y_stage[BT * H_DIM];

    const int tid   = threadIdx.x;
    const int lane  = tid & 31;
    const int wave  = tid >> 5;          // 0..15, owns H-cols [wave*16, wave*16+16)
    const int hi    = (lane >> 4) & 1;   // lane half -> K/M sub-block select
    const int lr    = lane & 15;         // A row / B,C,D column within tile
    const int bbase = blockIdx.x * BT;
    const int hcol  = wave * 16 + lr;    // this lane's H column (0..255)
    const int Mbase = hi * 8;            // C/D rows: Mbase + r, r = 0..7

    // per-gate weight row base pointers for this lane's column (loop-invariant)
    const __bf16* wxg[4];
    const __bf16* whg[4];
    float bias4[4];
    #pragma unroll
    for (int g = 0; g < 4; ++g) {
        wxg[g]   = WxT + (size_t)(g * H_DIM + hcol) * D_DIM;
        whg[g]   = WhT + (size_t)(g * H_DIM + hcol) * H_DIM;
        bias4[g] = bias[g * H_DIM + hcol];
    }

    // init c from c0 (matches C/D accumulator layout)
    float c[8];
    #pragma unroll
    for (int r = 0; r < 8; ++r)
        c[r] = c0[(size_t)(bbase + Mbase + r) * H_DIM + hcol];

    // init h_lds from h0 (bf16)
    if (tid < 256) {
        int row = tid >> 4, chunk = tid & 15;
        const float* hp = h0 + (size_t)(bbase + row) * H_DIM + chunk * 16;
        __bf16* dp = h_lds + row * HPITCH + chunk * 16;
        #pragma unroll
        for (int j = 0; j < 16; ++j) dp[j] = f2bf(hp[j]);
    }
    __syncthreads();

    float nh[8];
    for (int t = 0; t < T_DIM; ++t) {
        // ---- stage x tile (fp32 -> bf16) into LDS: 512 thr x 4 elems ----
        {
            int row = tid >> 5;              // 0..15
            int col = lane * 4;              // 0..124
            const f32x4v* sp =
                (const f32x4v*)(obs + ((size_t)t * B_DIM + bbase + row) * D_DIM + col);
            f32x4v xv = __builtin_nontemporal_load(sp);
            bf16x4v bv;
            bv.x = f2bf(xv.x); bv.y = f2bf(xv.y); bv.z = f2bf(xv.z); bv.w = f2bf(xv.w);
            *(bf16x4v*)(x_lds + row * XPITCH + col) = bv;
        }
        // ---- reset: zero h rows in LDS ----
        if (tid < 256) {
            int row = tid >> 4, chunk = tid & 15;
            if (reset[t * B_DIM + bbase + row]) {
                int4 z = {0, 0, 0, 0};
                int4* dp = (int4*)(h_lds + row * HPITCH + chunk * 16);
                dp[0] = z; dp[1] = z;
            }
        }
        // ---- reset: zero c in registers ----
        {
            const int* rp = reset + t * B_DIM + bbase + Mbase;   // 32B aligned
            int4 r0 = *(const int4*)rp;
            int4 r1 = *(const int4*)(rp + 4);
            if (r0.x) c[0] = 0.f;  if (r0.y) c[1] = 0.f;
            if (r0.z) c[2] = 0.f;  if (r0.w) c[3] = 0.f;
            if (r1.x) c[4] = 0.f;  if (r1.y) c[5] = 0.f;
            if (r1.z) c[6] = 0.f;  if (r1.w) c[7] = 0.f;
        }
        __syncthreads();

        // ---- accumulators start at bias ----
        f32x8 acc[4];
        #pragma unroll
        for (int g = 0; g < 4; ++g) {
            #pragma unroll
            for (int r = 0; r < 8; ++r) acc[g][r] = bias4[g];
        }

        // ---- x @ Wx : 4 K-tiles of 32 over D ----
        #pragma unroll 2
        for (int kt = 0; kt < 4; ++kt) {
            AFrag a;
            const __bf16* ap = x_lds + lr * XPITCH + kt * 32 + hi * 8;
            a.q[0] = *(const int4*)ap;          // K = kt*32 + hi*8 + 0..7
            a.q[1] = *(const int4*)(ap + 16);   // K = kt*32 + hi*8 + 16..23
            int KB = kt * 32 + hi * 8;
            #pragma unroll
            for (int g = 0; g < 4; ++g) {
                AFrag bb;
                const __bf16* bp = wxg[g] + KB;
                bb.q[0] = *(const int4*)bp;
                bb.q[1] = *(const int4*)(bp + 16);
                acc[g] = __builtin_amdgcn_wmma_f32_16x16x32_bf16(
                    false, a.v, false, bb.v, (short)0, acc[g], false, false);
            }
        }

        // ---- h @ Wh : 8 K-tiles of 32 over H ----
        #pragma unroll 2
        for (int kt = 0; kt < 8; ++kt) {
            AFrag a;
            const __bf16* ap = h_lds + lr * HPITCH + kt * 32 + hi * 8;
            a.q[0] = *(const int4*)ap;
            a.q[1] = *(const int4*)(ap + 16);
            int KB = kt * 32 + hi * 8;
            #pragma unroll
            for (int g = 0; g < 4; ++g) {
                AFrag bb;
                const __bf16* bp = whg[g] + KB;
                bb.q[0] = *(const int4*)bp;
                bb.q[1] = *(const int4*)(bp + 16);
                acc[g] = __builtin_amdgcn_wmma_f32_16x16x32_bf16(
                    false, a.v, false, bb.v, (short)0, acc[g], false, false);
            }
        }

        // ---- gates + state update (fp32) ----
        #pragma unroll
        for (int r = 0; r < 8; ++r) {
            float iv = sigmoidf_(acc[0][r]);
            float fv = sigmoidf_(acc[1][r]);
            float gv = tanhf_(acc[2][r]);
            float ov = sigmoidf_(acc[3][r]);
            float nc = fv * c[r] + iv * gv;
            c[r] = nc;
            nh[r] = ov * tanhf_(nc);
        }
        __syncthreads();   // all LDS reads done before overwrite

        #pragma unroll
        for (int r = 0; r < 8; ++r) {
            int M = Mbase + r;
            h_lds[M * HPITCH + hcol] = f2bf(nh[r]);
            y_stage[M * H_DIM + hcol] = nh[r];
        }
        __syncthreads();   // h/y tiles visible

        // coalesced non-temporal ys store: wave = row, lane covers 8 floats
        {
            int row = wave;
            int col = lane * 8;
            const float* sp = y_stage + row * H_DIM + col;
            float* dp = out + ((size_t)t * B_DIM + bbase + row) * H_DIM + col;
            __builtin_nontemporal_store(*(const f32x4v*)sp,       (f32x4v*)dp);
            __builtin_nontemporal_store(*(const f32x4v*)(sp + 4), (f32x4v*)(dp + 4));
        }
    }

    // ---- final cf, hf ----
    {
        float* cf = out + (size_t)T_DIM * B_DIM * H_DIM;
        float* hf = cf + (size_t)B_DIM * H_DIM;
        #pragma unroll
        for (int r = 0; r < 8; ++r) {
            int b = bbase + Mbase + r;
            cf[(size_t)b * H_DIM + hcol] = c[r];
            hf[(size_t)b * H_DIM + hcol] = nh[r];
        }
    }
}

extern "C" void kernel_launch(void* const* d_in, const int* in_sizes, int n_in,
                              void* d_out, int out_size, void* d_ws, size_t ws_size,
                              hipStream_t stream) {
    const float* obs  = (const float*)d_in[0];
    const int*   rst  = (const int*)d_in[1];
    const float* c0   = (const float*)d_in[2];
    const float* h0   = (const float*)d_in[3];
    const float* Wx   = (const float*)d_in[4];
    const float* Wh   = (const float*)d_in[5];
    const float* bias = (const float*)d_in[6];
    float* out = (float*)d_out;

    __bf16* WxT = (__bf16*)d_ws;                 // 4H x D  bf16 = 256 KB
    __bf16* WhT = WxT + (size_t)FH_DIM * D_DIM;  // 4H x H  bf16 = 512 KB

    int total = FH_DIM * D_DIM + FH_DIM * H_DIM;
    cvt_weights<<<(total + 255) / 256, 256, 0, stream>>>(Wx, Wh, WxT, WhT);
    lstm_scan<<<B_DIM / BT, 512, 0, stream>>>(obs, rst, c0, h0, bias, WxT, WhT, out);
}
